// BilinearAttention_15298673508534
// MI455X (gfx1250) — matrix operand order repaired and checked by reading, not compiled, not run on hardware
//
#include <hip/hip_runtime.h>
#include <math.h>

// ---------- types ----------
typedef __attribute__((ext_vector_type(16))) __bf16 v16bf;
typedef __attribute__((ext_vector_type(8)))  float  v8f;
typedef int v4i __attribute__((vector_size(16)));

union Frag16 { uint4 u[2]; v16bf bf; };

#define AS1 __attribute__((address_space(1)))
#define AS3 __attribute__((address_space(3)))

#if __has_builtin(__builtin_amdgcn_global_load_async_to_lds_b128)
#define HAVE_ASYNC_LDS 1
#else
#define HAVE_ASYNC_LDS 0
#endif

static __device__ __forceinline__ void wait_async0() {
#if __has_builtin(__builtin_amdgcn_s_wait_asynccnt)
    __builtin_amdgcn_s_wait_asynccnt(0);
#else
    asm volatile("s_wait_asynccnt 0x0" ::: "memory");
#endif
}

static __device__ __forceinline__ unsigned short f2bf(float f) {
    unsigned int u = __float_as_uint(f);
    unsigned int r = u + 0x7FFFu + ((u >> 16) & 1u);   // round-to-nearest-even
    return (unsigned short)(r >> 16);
}

// ---------- f32 -> bf16 convert (optionally strided destination) ----------
__global__ __launch_bounds__(256) void k_cvt(const float* __restrict__ src,
                                             unsigned short* __restrict__ dst,
                                             int cols, int ldd) {
    size_t idx = (size_t)blockIdx.x * 256 + threadIdx.x;
    int row = (int)(idx / cols);
    int col = (int)(idx % cols);
    dst[(size_t)row * ldd + col] = f2bf(src[idx]);
}

// ---------- f32 [R,C] -> bf16 [C,R] transpose+convert ----------
__global__ __launch_bounds__(256) void k_tr_cvt(const float* __restrict__ src,
                                                unsigned short* __restrict__ dst,
                                                int R, int C) {
    __shared__ float t[32][33];
    int c0 = blockIdx.x * 32, r0 = blockIdx.y * 32;
    int tx = threadIdx.x, ty = threadIdx.y;
#pragma unroll
    for (int r = 0; r < 4; r++)
        t[ty + 8 * r][tx] = src[(size_t)(r0 + ty + 8 * r) * C + c0 + tx];
    __syncthreads();
#pragma unroll
    for (int r = 0; r < 4; r++)
        dst[(size_t)(c0 + ty + 8 * r) * R + r0 + tx] = f2bf(t[tx][ty + 8 * r]);
}

// ---------- column softmax stats, stage 1: partial (max, sum-exp) per 256-row chunk ----------
__global__ __launch_bounds__(256) void k_colstats1(const float* __restrict__ S,
                                                   float* __restrict__ pm,
                                                   float* __restrict__ ps,
                                                   int Bdim) {
    int j = blockIdx.x * 256 + threadIdx.x;
    int i0 = blockIdx.y * 256;
    float m = -3.402823466e38f;
    for (int i = 0; i < 256; i++)
        m = fmaxf(m, S[(size_t)(i0 + i) * Bdim + j]);
    float s = 0.f;
    for (int i = 0; i < 256; i++)
        s += __expf(S[(size_t)(i0 + i) * Bdim + j] - m);
    pm[(size_t)blockIdx.y * Bdim + j] = m;
    ps[(size_t)blockIdx.y * Bdim + j] = s;
}

// ---------- column softmax stats, stage 2: combine partials ----------
__global__ __launch_bounds__(256) void k_colstats2(const float* __restrict__ pm,
                                                   const float* __restrict__ ps,
                                                   float* __restrict__ colm,
                                                   float* __restrict__ colr,
                                                   int Bdim, int chunks) {
    int j = blockIdx.x * 256 + threadIdx.x;
    float m = -3.402823466e38f;
    for (int c = 0; c < chunks; c++)
        m = fmaxf(m, pm[(size_t)c * Bdim + j]);
    float s = 0.f;
    for (int c = 0; c < chunks; c++)
        s += ps[(size_t)c * Bdim + j] * __expf(pm[(size_t)c * Bdim + j] - m);
    colm[j] = m;
    colr[j] = 1.0f / s;
}

// ---------- normalize + transpose: attnT[j][i] = bf16(exp(S[i][j]-m[j])/s[j]) ----------
__global__ __launch_bounds__(256) void k_attn_tr(const float* __restrict__ S,
                                                 const float* __restrict__ colm,
                                                 const float* __restrict__ colr,
                                                 unsigned short* __restrict__ attnT,
                                                 int Bdim) {
    __shared__ float t[32][33];
    int j0 = blockIdx.x * 32, i0 = blockIdx.y * 32;
    int tx = threadIdx.x, ty = threadIdx.y;
#pragma unroll
    for (int r = 0; r < 4; r++)
        t[ty + 8 * r][tx] = S[(size_t)(i0 + ty + 8 * r) * Bdim + j0 + tx];
    __syncthreads();
#pragma unroll
    for (int r = 0; r < 4; r++) {
        int j = j0 + ty + 8 * r;
        float v = __expf(t[tx][ty + 8 * r] - colm[j]) * colr[j];
        attnT[(size_t)j * Bdim + i0 + tx] = f2bf(v);
    }
}

// ---------- bf16 WMMA GEMM, C[M,N] = A[M,K] * B[N,K]^T (+bias, +tanh) ----------
// Block: 256 threads = 8 wave32s, block tile 128x128, wave tile 32x64, K-step 32.
template <bool OUT_BF16, bool HAS_BIAS, bool DO_TANH>
__global__ __launch_bounds__(256) void k_gemm_nt(const unsigned short* __restrict__ Ag,
                                                 const unsigned short* __restrict__ Bg,
                                                 const float* __restrict__ bias,
                                                 void* __restrict__ Cg,
                                                 int K, int lda, int ldb, int ldc) {
    __shared__ __align__(16) unsigned short sA[2][128 * 32];
    __shared__ __align__(16) unsigned short sB[2][128 * 32];

    const int tid   = threadIdx.x;
    const int lane  = tid & 31;
    const int wave  = tid >> 5;
    const int waveM = wave & 3;   // 0..3 -> 32-row strips
    const int waveN = wave >> 2;  // 0..1 -> 64-col strips
    const int m0 = blockIdx.y * 128;
    const int n0 = blockIdx.x * 128;

    v8f acc[2][4];
#pragma unroll
    for (int r = 0; r < 2; r++)
#pragma unroll
        for (int t = 0; t < 4; t++)
#pragma unroll
            for (int e = 0; e < 8; e++) acc[r][t][e] = 0.f;

    auto stage = [&](int buf, int kb) {
        const int k0 = kb * 32;
#pragma unroll
        for (int s = 0; s < 2; s++) {
            int chunk = tid + 256 * s;        // 512 chunks of 16B per 8KB tile
            int row = chunk >> 2;
            int co  = (chunk & 3) << 3;       // element offset, 8 bf16 per chunk
#if HAVE_ASYNC_LDS
            __builtin_amdgcn_global_load_async_to_lds_b128(
                (AS1 v4i*)&Ag[(size_t)(m0 + row) * lda + k0 + co],
                (AS3 v4i*)&sA[buf][row * 32 + co], 0, 0);
            __builtin_amdgcn_global_load_async_to_lds_b128(
                (AS1 v4i*)&Bg[(size_t)(n0 + row) * ldb + k0 + co],
                (AS3 v4i*)&sB[buf][row * 32 + co], 0, 0);
#else
            *(uint4*)&sA[buf][row * 32 + co] =
                *(const uint4*)&Ag[(size_t)(m0 + row) * lda + k0 + co];
            *(uint4*)&sB[buf][row * 32 + co] =
                *(const uint4*)&Bg[(size_t)(n0 + row) * ldb + k0 + co];
#endif
        }
    };

    stage(0, 0);
#if HAVE_ASYNC_LDS
    wait_async0();
#endif
    __syncthreads();

    const int ml  = lane & 15;
    const int ksa = (lane >> 4) << 4;   // A: K start 0 or 16 (contiguous K, matches B)
    const int ksb = (lane >> 4) << 4;   // B: K start 0 or 16 (contiguous K, matches A)

    const int KB = K >> 5;
    for (int kb = 0; kb < KB; ++kb) {
        const int cur = kb & 1;
        if (kb + 1 < KB) stage(cur ^ 1, kb + 1);

        const unsigned short* a = sA[cur];
        const unsigned short* b = sB[cur];

        Frag16 af[2];
#pragma unroll
        for (int r = 0; r < 2; r++) {
            int m = waveM * 32 + r * 16 + ml;
            af[r].u[0] = *(const uint4*)&a[m * 32 + ksa];        // K ks..ks+7
            af[r].u[1] = *(const uint4*)&a[m * 32 + ksa + 8];    // K ks+8..ks+15
        }
        Frag16 bf[4];
#pragma unroll
        for (int t = 0; t < 4; t++) {
            int c = waveN * 64 + t * 16 + ml;
            bf[t].u[0] = *(const uint4*)&b[c * 32 + ksb];        // K ks..ks+7
            bf[t].u[1] = *(const uint4*)&b[c * 32 + ksb + 8];    // K ks+8..ks+15
        }
#pragma unroll
        for (int r = 0; r < 2; r++)
#pragma unroll
            for (int t = 0; t < 4; t++)
                acc[r][t] = __builtin_amdgcn_wmma_f32_16x16x32_bf16(
                    false, af[r].bf, false, bf[t].bf, (short)0, acc[r][t], false, false);

#if HAVE_ASYNC_LDS
        wait_async0();
#endif
        __syncthreads();
    }

    // Epilogue. C/D layout: VGPR p -> M = p + (lane>=16 ? 8 : 0), N = lane&15.
    const int mo = (lane >> 4) << 3;
#pragma unroll
    for (int r = 0; r < 2; r++) {
#pragma unroll
        for (int t = 0; t < 4; t++) {
            const int col  = n0 + waveN * 64 + t * 16 + ml;
            const int rowb = m0 + waveM * 32 + r * 16 + mo;
            float bv = 0.f;
            if (HAS_BIAS) bv = bias[col];
#pragma unroll
            for (int p = 0; p < 8; p++) {
                float x = acc[r][t][p] + bv;
                if (DO_TANH) x = tanhf(x);
                size_t off = (size_t)(rowb + p) * ldc + col;
                if (OUT_BF16) ((unsigned short*)Cg)[off] = f2bf(x);
                else          ((float*)Cg)[off] = x;
            }
        }
    }
}

// ---------- launch ----------
extern "C" void kernel_launch(void* const* d_in, const int* in_sizes, int n_in,
                              void* d_out, int out_size, void* d_ws, size_t ws_size,
                              hipStream_t stream) {
    (void)in_sizes; (void)n_in; (void)out_size; (void)ws_size;
    const int Bdim = 8192, H = 1024, A = 1024;
    const int CHUNKS = 32;   // 8192 / 256 rows per chunk

    const float* attendee = (const float*)d_in[0];   // [B,H]
    const float* attender = (const float*)d_in[1];   // [B,H]
    const float* W_score  = (const float*)d_in[2];   // [H,H]
    const float* b_score  = (const float*)d_in[3];   // [H]
    const float* W_out    = (const float*)d_in[4];   // [A,2H]
    const float* b_out    = (const float*)d_in[5];   // [A]
    float* out = (float*)d_out;                      // [B,A]

    unsigned char* w = (unsigned char*)d_ws;
    size_t off = 0;
    auto alloc = [&](size_t bytes) {
        void* p = w + off;
        off = (off + bytes + 255) & ~(size_t)255;
        return p;
    };
    unsigned short* aee_bf  = (unsigned short*)alloc((size_t)Bdim * H * 2);      // attendee bf16
    unsigned short* aer_bf  = (unsigned short*)alloc((size_t)Bdim * H * 2);      // attender bf16
    unsigned short* wsc_bf  = (unsigned short*)alloc((size_t)H * H * 2);         // W_score bf16 [N,K]
    unsigned short* wo_bf   = (unsigned short*)alloc((size_t)A * 2 * H * 2);     // W_out bf16 [N,K]
    unsigned short* aeeT_bf = (unsigned short*)alloc((size_t)H * Bdim * 2);      // attendee^T bf16
    unsigned short* P_bf    = (unsigned short*)alloc((size_t)Bdim * H * 2);      // P bf16
    unsigned short* concat  = (unsigned short*)alloc((size_t)Bdim * 2 * H * 2);  // [attender|ctx] bf16
    float*          colm    = (float*)alloc((size_t)Bdim * 4);
    float*          colr    = (float*)alloc((size_t)Bdim * 4);
    float*          pm      = (float*)alloc((size_t)CHUNKS * Bdim * 4);
    float*          ps      = (float*)alloc((size_t)CHUNKS * Bdim * 4);
    unsigned short* attnT   = (unsigned short*)alloc((size_t)Bdim * Bdim * 2);   // attn^T bf16
    float*          S       = (float*)alloc((size_t)Bdim * Bdim * 4);            // scores f32

    dim3 blk256(256);
    // convert inputs to bf16
    k_cvt<<<dim3((Bdim * H) / 256), blk256, 0, stream>>>(attendee, aee_bf, H, H);
    k_cvt<<<dim3((Bdim * H) / 256), blk256, 0, stream>>>(attender, aer_bf, H, H);
    k_cvt<<<dim3((Bdim * H) / 256), blk256, 0, stream>>>(attender, concat, H, 2 * H); // concat[:, :H]
    k_cvt<<<dim3((H * H) / 256),    blk256, 0, stream>>>(W_score, wsc_bf, H, H);
    k_cvt<<<dim3((A * 2 * H) / 256), blk256, 0, stream>>>(W_out, wo_bf, 2 * H, 2 * H);
    // attendee^T bf16 (B operand for GEMM3)
    k_tr_cvt<<<dim3(H / 32, Bdim / 32), dim3(32, 8), 0, stream>>>(attendee, aeeT_bf, Bdim, H);

    // GEMM1: P = attendee @ W_score^T + b_score      [B,H] bf16
    k_gemm_nt<true, true, false><<<dim3(H / 128, Bdim / 128), blk256, 0, stream>>>(
        aee_bf, wsc_bf, b_score, P_bf, H, H, H, H);

    // GEMM2: S = P @ attender^T                      [B,B] f32
    k_gemm_nt<false, false, false><<<dim3(Bdim / 128, Bdim / 128), blk256, 0, stream>>>(
        P_bf, aer_bf, nullptr, S, H, H, H, Bdim);

    // column softmax stats (two-stage for memory-level parallelism)
    k_colstats1<<<dim3(Bdim / 256, CHUNKS), blk256, 0, stream>>>(S, pm, ps, Bdim);
    k_colstats2<<<dim3(Bdim / 256), blk256, 0, stream>>>(pm, ps, colm, colr, Bdim, CHUNKS);
    // attn^T = softmax columns, transposed, bf16
    k_attn_tr<<<dim3(Bdim / 32, Bdim / 32), dim3(32, 8), 0, stream>>>(S, colm, colr, attnT, Bdim);

    // GEMM3: ctx = attn^T @ attendee  -> concat[:, H:2H] bf16
    k_gemm_nt<true, false, false><<<dim3(H / 128, Bdim / 128), blk256, 0, stream>>>(
        attnT, aeeT_bf, nullptr, concat + H, Bdim, Bdim, Bdim, 2 * H);

    // GEMM4: out = tanh(concat @ W_out^T + b_out)    [B,A] f32
    k_gemm_nt<false, true, true><<<dim3(A / 128, Bdim / 128), blk256, 0, stream>>>(
        concat, wo_bf, b_out, out, 2 * H, 2 * H, 2 * H, A);
}
